// Attention_72438918414857
// MI455X (gfx1250) — compile-verified, matrix operand order checked by reading, not run on hardware
//
#include <hip/hip_runtime.h>

// ---- CDNA5 wave32 WMMA types ----
typedef __attribute__((ext_vector_type(16))) __bf16        v16bf;
typedef __attribute__((ext_vector_type(8)))  float         v8f;
typedef __attribute__((ext_vector_type(2)))  unsigned int  u32x2;
typedef __attribute__((ext_vector_type(4)))  unsigned int  u32x4;
typedef __attribute__((ext_vector_type(8)))  unsigned int  u32x8;
typedef __attribute__((ext_vector_type(4)))  float         f32x4;

// fp32 -> bf16 (round-to-nearest-even), bit-level
static __device__ __forceinline__ unsigned int bfbits(float f) {
  unsigned int u = __builtin_bit_cast(unsigned int, f);
  u += 0x7FFFu + ((u >> 16) & 1u);
  return u >> 16;
}
static __device__ __forceinline__ unsigned short f2bf(float f) {
  return (unsigned short)bfbits(f);
}
static __device__ __forceinline__ unsigned int pack2(float a, float b) {
  return bfbits(a) | (bfbits(b) << 16);
}
static __device__ __forceinline__ f32x4 ldf4(const float* p) {
  return *reinterpret_cast<const f32x4*>(p);
}
static __device__ __forceinline__ u32x4 ldu4(const unsigned short* p) {
  return *reinterpret_cast<const u32x4*>(p);
}

// 16x32 bf16 A/B fragment from bf16 memory (two b128 strips: K+0..7, K+16..23).
// Per-lane element i maps to K = koff + (i<8 ? i : i+8), koff = (lane>>4)*8.
static __device__ __forceinline__ v16bf frag_from_bf16(const unsigned short* p) {
  u32x4 lo = ldu4(p), hi = ldu4(p + 16);
  u32x8 t = __builtin_shufflevector(lo, hi, 0, 1, 2, 3, 4, 5, 6, 7);
  return __builtin_bit_cast(v16bf, t);
}
static __device__ __forceinline__ v8f wmma_bf16(v16bf a, v16bf b, v8f c) {
  return __builtin_amdgcn_wmma_f32_16x16x32_bf16(false, a, false, b, (short)0, c,
                                                 false, false);
}

// Async copy of one 16-byte chunk: global -> LDS (tracked by ASYNCcnt).
static __device__ __forceinline__ void async_g2l_b128(unsigned lds_addr,
                                                      unsigned long long gaddr) {
  asm volatile("global_load_async_to_lds_b128 %0, %1, off"
               :: "v"(lds_addr), "v"(gaddr) : "memory");
}
static __device__ __forceinline__ void wait_asynccnt0() {
  asm volatile("s_wait_asynccnt 0x0" ::: "memory");
}

// ============================================================================
// Kernel 0: one-shot fp32 -> bf16 conversion (memory bound, runs once per array)
// ============================================================================
__global__ __launch_bounds__(256) void cvt_bf16(const float* __restrict__ in,
                                                unsigned short* __restrict__ out,
                                                int n4) {
  const int i = blockIdx.x * 256 + threadIdx.x;
  if (i < n4) {
    f32x4 f = ldf4(in + (size_t)i * 4);
    u32x2 p;
    p[0] = pack2(f[0], f[1]);
    p[1] = pack2(f[2], f[3]);
    *reinterpret_cast<u32x2*>(out + (size_t)i * 4) = p;
  }
}

// ============================================================================
// Kernel 1: QKV projection  [8192,768]bf16 x [2304,768]bf16^T + bias(f32)
// Block = 8 waves = 2 M-tiles x 4 N-groups (32 rows x 256 cols of output).
// The 32x768 A panel (48 KB, contiguous) is async-copied to LDS once per block;
// waves read A fragments via ds_load_b128 and stream B from global (pipelined).
// Writes q (pre-scaled by D^-0.5), k as bf16 [b,h,n,d]; v as bf16 [b,h,d,n].
// ============================================================================
__global__ __launch_bounds__(256) void qkv_gemm(
    const unsigned short* __restrict__ xbf, const unsigned short* __restrict__ wbf,
    const float* __restrict__ bias, unsigned short* __restrict__ qws,
    unsigned short* __restrict__ kws, unsigned short* __restrict__ vtws) {
  __shared__ __align__(16) unsigned short sA[32 * 768];  // 48 KB

  const int mgrp = blockIdx.x / 9, ng = blockIdx.x % 9;  // 256 x 9 blocks
  const int wave = threadIdx.x >> 5, lane = threadIdx.x & 31;
  const int mt_local = wave >> 2;
  const int nbase = (ng * 4 + (wave & 3)) << 6;
  const int mbase = (mgrp << 5) + (mt_local << 4);
  const int lrow = lane & 15, lhi = lane >> 4, koff = lhi << 3;

  // ---- async stage of A panel: 32 rows x 768 cols = 49152 B, contiguous ----
  {
    const unsigned long long gp =
        (unsigned long long)(uintptr_t)(xbf + (size_t)(mgrp << 5) * 768);
    const unsigned lbase = (unsigned)(uintptr_t)(&sA[0]);
#pragma unroll
    for (int c = 0; c < 12; ++c) {
      const int idx = threadIdx.x + (c << 8);
      async_g2l_b128(lbase + idx * 16, gp + (unsigned long long)idx * 16);
    }
    wait_asynccnt0();
  }
  __syncthreads();

  const unsigned short* ap = sA + (size_t)((mt_local << 4) + lrow) * 768 + koff;
  const unsigned short* bp = wbf + (size_t)(nbase + lrow) * 768 + koff;

  v8f acc0 = {}, acc1 = {}, acc2 = {}, acc3 = {};
  v16bf b0 = frag_from_bf16(bp);
  v16bf b1 = frag_from_bf16(bp + 16 * 768);
  v16bf b2 = frag_from_bf16(bp + 32 * 768);
  v16bf b3 = frag_from_bf16(bp + 48 * 768);
#pragma unroll 2
  for (int k0 = 32; k0 < 768; k0 += 32) {
    v16bf b0n = frag_from_bf16(bp + k0);
    v16bf b1n = frag_from_bf16(bp + 16 * 768 + k0);
    v16bf b2n = frag_from_bf16(bp + 32 * 768 + k0);
    v16bf b3n = frag_from_bf16(bp + 48 * 768 + k0);
    v16bf a = frag_from_bf16(ap + (k0 - 32));  // LDS
    acc0 = wmma_bf16(a, b0, acc0);
    acc1 = wmma_bf16(a, b1, acc1);
    acc2 = wmma_bf16(a, b2, acc2);
    acc3 = wmma_bf16(a, b3, acc3);
    b0 = b0n; b1 = b1n; b2 = b2n; b3 = b3n;
  }
  {
    v16bf a = frag_from_bf16(ap + 736);
    acc0 = wmma_bf16(a, b0, acc0);
    acc1 = wmma_bf16(a, b1, acc1);
    acc2 = wmma_bf16(a, b2, acc2);
    acc3 = wmma_bf16(a, b3, acc3);
  }

  v8f accs[4] = {acc0, acc1, acc2, acc3};
#pragma unroll
  for (int t = 0; t < 4; ++t) {
    v8f c = accs[t];
    const int col = nbase + (t << 4) + lrow;
    const int which = col / 768, cc = col % 768;  // 0=q 1=k 2=v
    const int h = cc >> 6, d = cc & 63;
    const float bv = bias[col];
#pragma unroll
    for (int r = 0; r < 8; ++r) {
      const int m = mbase + r + (lhi << 3);
      const int bidx = m >> 10, n = m & 1023;
      const float v = c[r] + bv;
      const size_t head = (size_t)(bidx * 12 + h);
      if (which == 0)      qws[(head * 1024 + n) * 64 + d] = f2bf(v * 0.125f);
      else if (which == 1) kws[(head * 1024 + n) * 64 + d] = f2bf(v);
      else                 vtws[(head * 64 + d) * 1024 + n] = f2bf(v);
    }
  }
}

// ============================================================================
// Kernel 2: fused attention.  Block = (b, h, 16-row q tile), 4 waves.
// scores(f32, LDS) -> softmax -> +static_a -> probs(bf16, LDS) -> P@V -> bf16 ws
// ============================================================================
__global__ __launch_bounds__(128) void attn_fused(
    const unsigned short* __restrict__ qws, const unsigned short* __restrict__ kws,
    const unsigned short* __restrict__ vtws, const float* __restrict__ static_a,
    unsigned short* __restrict__ aws) {
  extern __shared__ char smem[];
  float* sc = reinterpret_cast<float*>(smem);                                   // 16x1024 f32
  unsigned short* pr = reinterpret_cast<unsigned short*>(smem + 16 * 1024 * 4); // 16x1024 bf16
  float* red = reinterpret_cast<float*>(smem + 16 * 1024 * 4 + 16 * 1024 * 2);  // 16x8 f32

  const int nt = blockIdx.x, h = blockIdx.y, bb = blockIdx.z;
  const int wave = threadIdx.x >> 5, lane = threadIdx.x & 31;
  const int lrow = lane & 15, lhi = lane >> 4, koff = lhi << 3;
  const size_t head = (size_t)(bb * 12 + h) * (1024 * 64);

  // ---- scores = q @ k^T (scale pre-folded into q); each wave covers 256 cols
  const unsigned short* qp = qws + head + (size_t)((nt << 4) + lrow) * 64 + koff;
  v16bf a0 = frag_from_bf16(qp);       // d = 0..31
  v16bf a1 = frag_from_bf16(qp + 32);  // d = 32..63
  const unsigned short* kp = kws + head + (size_t)((wave << 8) + lrow) * 64 + koff;
  v16bf b0 = frag_from_bf16(kp);
  v16bf b1 = frag_from_bf16(kp + 32);
  for (int j = 0; j < 15; ++j) {
    const unsigned short* kpn = kp + (size_t)(j + 1) * 1024;  // next 16-col tile
    v16bf b0n = frag_from_bf16(kpn);
    v16bf b1n = frag_from_bf16(kpn + 32);
    v8f c = {};
    c = wmma_bf16(a0, b0, c);
    c = wmma_bf16(a1, b1, c);
    const int colg = (wave << 8) + (j << 4) + lrow;
#pragma unroll
    for (int r = 0; r < 8; ++r) sc[(r + (lhi << 3)) * 1024 + colg] = c[r];
    b0 = b0n; b1 = b1n;
  }
  {
    v8f c = {};
    c = wmma_bf16(a0, b0, c);
    c = wmma_bf16(a1, b1, c);
    const int colg = (wave << 8) + (15 << 4) + lrow;
#pragma unroll
    for (int r = 0; r < 8; ++r) sc[(r + (lhi << 3)) * 1024 + colg] = c[r];
  }
  __syncthreads();

  // ---- row softmax (16 rows, 8 threads/row scanning 128 elems each)
  const int rr = threadIdx.x >> 3, ch = threadIdx.x & 7;
  float* rowp = sc + rr * 1024 + (ch << 7);
  float mx = -3.402823466e38f;
  for (int i = 0; i < 128; ++i) mx = fmaxf(mx, rowp[i]);
  red[(rr << 3) + ch] = mx;
  __syncthreads();
  float rowmax = red[rr << 3];
#pragma unroll
  for (int t = 1; t < 8; ++t) rowmax = fmaxf(rowmax, red[(rr << 3) + t]);
  __syncthreads();
  float s = 0.f;
  for (int i = 0; i < 128; ++i) {
    float e = __expf(rowp[i] - rowmax);
    rowp[i] = e;
    s += e;
  }
  red[(rr << 3) + ch] = s;
  __syncthreads();
  float rs = 0.f;
#pragma unroll
  for (int t = 0; t < 8; ++t) rs += red[(rr << 3) + t];
  const float inv = 1.0f / rs;
  // static bias added AFTER softmax (no renorm), matching the reference
  const float* sap = static_a + ((size_t)h * 1024 + (nt << 4) + rr) * 1024 + (ch << 7);
  unsigned short* prp = pr + rr * 1024 + (ch << 7);
  for (int i = 0; i < 128; ++i) prp[i] = f2bf(rowp[i] * inv + sap[i]);
  __syncthreads();

  // ---- out[16 x 64] = probs @ V; wave owns d-columns [16w, 16w+16)
  const int d = (wave << 4) + lrow;
  const unsigned short* vp = vtws + head + (size_t)d * 1024;  // contiguous in K=m
  const unsigned short* prow = pr + lrow * 1024;
  v8f c = {};
  v16bf pa = frag_from_bf16(prow + koff);   // ds_load_b128 from LDS
  v16bf vb = frag_from_bf16(vp + koff);
  for (int it = 1; it < 32; ++it) {
    const int kb = (it << 5) + koff;
    v16bf pan = frag_from_bf16(prow + kb);
    v16bf vbn = frag_from_bf16(vp + kb);
    c = wmma_bf16(pa, vb, c);
    pa = pan; vb = vbn;
  }
  c = wmma_bf16(pa, vb, c);
#pragma unroll
  for (int r = 0; r < 8; ++r) {
    const int n = (nt << 4) + r + (lhi << 3);
    aws[(size_t)(bb * 1024 + n) * 768 + (h << 6) + d] = f2bf(c[r]);
  }
}

// ============================================================================
// Kernel 3: output projection [8192,768]bf16 x [768,768]bf16^T + bias -> fp32
// Same block structure as qkv_gemm: async A panel in LDS, B from global.
// ============================================================================
__global__ __launch_bounds__(256) void proj_gemm(
    const unsigned short* __restrict__ aws, const unsigned short* __restrict__ wbf,
    const float* __restrict__ bias, float* __restrict__ out) {
  __shared__ __align__(16) unsigned short sA[32 * 768];  // 48 KB

  const int mgrp = blockIdx.x / 3, ng = blockIdx.x % 3;  // 256 x 3 blocks
  const int wave = threadIdx.x >> 5, lane = threadIdx.x & 31;
  const int mt_local = wave >> 2;
  const int nbase = (ng * 4 + (wave & 3)) << 6;
  const int mbase = (mgrp << 5) + (mt_local << 4);
  const int lrow = lane & 15, lhi = lane >> 4, koff = lhi << 3;

  {
    const unsigned long long gp =
        (unsigned long long)(uintptr_t)(aws + (size_t)(mgrp << 5) * 768);
    const unsigned lbase = (unsigned)(uintptr_t)(&sA[0]);
#pragma unroll
    for (int c = 0; c < 12; ++c) {
      const int idx = threadIdx.x + (c << 8);
      async_g2l_b128(lbase + idx * 16, gp + (unsigned long long)idx * 16);
    }
    wait_asynccnt0();
  }
  __syncthreads();

  const unsigned short* ap = sA + (size_t)((mt_local << 4) + lrow) * 768 + koff;
  const unsigned short* bp = wbf + (size_t)(nbase + lrow) * 768 + koff;

  v8f acc0 = {}, acc1 = {}, acc2 = {}, acc3 = {};
  v16bf b0 = frag_from_bf16(bp);
  v16bf b1 = frag_from_bf16(bp + 16 * 768);
  v16bf b2 = frag_from_bf16(bp + 32 * 768);
  v16bf b3 = frag_from_bf16(bp + 48 * 768);
#pragma unroll 2
  for (int k0 = 32; k0 < 768; k0 += 32) {
    v16bf b0n = frag_from_bf16(bp + k0);
    v16bf b1n = frag_from_bf16(bp + 16 * 768 + k0);
    v16bf b2n = frag_from_bf16(bp + 32 * 768 + k0);
    v16bf b3n = frag_from_bf16(bp + 48 * 768 + k0);
    v16bf a = frag_from_bf16(ap + (k0 - 32));  // LDS
    acc0 = wmma_bf16(a, b0, acc0);
    acc1 = wmma_bf16(a, b1, acc1);
    acc2 = wmma_bf16(a, b2, acc2);
    acc3 = wmma_bf16(a, b3, acc3);
    b0 = b0n; b1 = b1n; b2 = b2n; b3 = b3n;
  }
  {
    v16bf a = frag_from_bf16(ap + 736);
    acc0 = wmma_bf16(a, b0, acc0);
    acc1 = wmma_bf16(a, b1, acc1);
    acc2 = wmma_bf16(a, b2, acc2);
    acc3 = wmma_bf16(a, b3, acc3);
  }

  v8f accs[4] = {acc0, acc1, acc2, acc3};
#pragma unroll
  for (int t = 0; t < 4; ++t) {
    v8f c = accs[t];
    const int col = nbase + (t << 4) + lrow;
    const float bv = bias[col];
#pragma unroll
    for (int r = 0; r < 8; ++r) {
      const int m = mbase + r + (lhi << 3);
      out[(size_t)m * 768 + col] = c[r] + bv;
    }
  }
}

extern "C" void kernel_launch(void* const* d_in, const int* in_sizes, int n_in,
                              void* d_out, int out_size, void* d_ws, size_t ws_size,
                              hipStream_t stream) {
  const float* x        = (const float*)d_in[0];
  const float* qkv_w    = (const float*)d_in[1];
  const float* qkv_b    = (const float*)d_in[2];
  const float* static_a = (const float*)d_in[3];
  const float* proj_w   = (const float*)d_in[4];
  const float* proj_b   = (const float*)d_in[5];
  float* out = (float*)d_out;

  // Workspace layout (bf16 elements):
  //   q/k/vT/attn_out : 4 x 6,291,456   (48 MiB)
  //   x_bf16          : 6,291,456       (12 MiB)
  //   qkv_w_bf16      : 1,769,472       (3.4 MiB)
  //   proj_w_bf16     : 589,824         (1.1 MiB)
  const size_t head_elems = (size_t)8 * 12 * 1024 * 64;  // 6,291,456
  unsigned short* qws  = (unsigned short*)d_ws;
  unsigned short* kws  = qws + head_elems;
  unsigned short* vtws = kws + head_elems;
  unsigned short* aws  = vtws + head_elems;
  unsigned short* xbf  = aws + head_elems;
  unsigned short* wqkv = xbf + head_elems;
  unsigned short* wprj = wqkv + (size_t)2304 * 768;

  // One-shot conversions
  const int n4_x = (8 * 1024 * 768) / 4;       // 1,572,864
  const int n4_wq = (2304 * 768) / 4;          //   442,368
  const int n4_wp = (768 * 768) / 4;           //   147,456
  cvt_bf16<<<dim3((n4_x + 255) / 256), dim3(256), 0, stream>>>(x, xbf, n4_x);
  cvt_bf16<<<dim3((n4_wq + 255) / 256), dim3(256), 0, stream>>>(qkv_w, wqkv, n4_wq);
  cvt_bf16<<<dim3((n4_wp + 255) / 256), dim3(256), 0, stream>>>(proj_w, wprj, n4_wp);

  // QKV: 256 M-groups x 9 N-groups of blocks (8 waves each)
  qkv_gemm<<<dim3(256 * 9), dim3(256), 0, stream>>>(xbf, wqkv, qkv_b, qws, kws, vtws);

  // Attention: one block per (16-row tile, head, batch); 98816 B dynamic LDS
  const size_t smem = 16 * 1024 * 4 + 16 * 1024 * 2 + 16 * 8 * 4;
  attn_fused<<<dim3(64, 12, 8), dim3(128), smem, stream>>>(qws, kws, vtws, static_a, aws);

  // Projection: 256 M-groups x 3 N-groups
  proj_gemm<<<dim3(256 * 3), dim3(256), 0, stream>>>(aws, wprj, proj_b, out);

  (void)in_sizes; (void)n_in; (void)out_size; (void)ws_size;
}
